// SAGE_4947802325594
// MI455X (gfx1250) — matrix operand order repaired and checked by reading, not compiled, not run on hardware
//
#include <hip/hip_runtime.h>
#include <hip/hip_bf16.h>

typedef __attribute__((ext_vector_type(16))) __bf16 v16bf;
typedef __attribute__((ext_vector_type(8)))  __bf16 v8bf;
typedef __attribute__((ext_vector_type(2)))  __bf16 v2bf;
typedef __attribute__((ext_vector_type(8)))  float  v8f;

#define FEAT 256  // IN_F == HID_F == 256: K dimension for every layer / gather width

// ---------------------------------------------------------------------------
// Edge-parallel segment-sum (mean numerator) + per-dst edge count.
// One wave per edge; lane covers 4 consecutive floats -> 2x global_load_b128,
// 8 atomic f32 adds (the atomics are the inherent cost of segment_sum).
// ---------------------------------------------------------------------------
__global__ __launch_bounds__(256)
void sage_scatter(const float* __restrict__ h,
                  const int* __restrict__ src,
                  const int* __restrict__ dst,
                  float* __restrict__ sum,
                  float* __restrict__ cnt,
                  int E) {
    int wid  = (int)((blockIdx.x * blockDim.x + threadIdx.x) >> 5);
    int lane = threadIdx.x & 31;
    if (wid >= E) return;
    int s = src[wid];
    int d = dst[wid];
    const float4* hp = (const float4*)(h + (size_t)s * FEAT);
    float* sp = sum + (size_t)d * FEAT;
#pragma unroll
    for (int j = 0; j < FEAT / 128; ++j) {
        float4 v = hp[lane + 32 * j];
        int o = 4 * (lane + 32 * j);
        atomicAdd(&sp[o + 0], v.x);
        atomicAdd(&sp[o + 1], v.y);
        atomicAdd(&sp[o + 2], v.z);
        atomicAdd(&sp[o + 3], v.w);
    }
    if (lane == 0) atomicAdd(&cnt[d], 1.0f);
}

// ---------------------------------------------------------------------------
// Fused SAGE GEMM:  Out = act( Aself@Wself + (Sum/max(Cnt,1))@Wneigh + bias )
//
// Block tile: 128 rows x 64 cols. 8 waves; wave w owns 16 rows, all 64 cols
// (4 accumulators). Per 32-wide k-step:
//   - B tiles (32x64, self + neigh) staged cooperatively (float2 loads)
//   - A tiles (16x32, self + neigh) staged wave-privately (float2 -> bf16x2)
//   - 8 x v_wmma_f32_16x16x32_bf16 per wave
// All global accesses are typed-base[int offset] so the backend emits the
// scalar-base + 32-bit voffset (saddr, scale_offset) addressing form.
// ---------------------------------------------------------------------------
__global__ __launch_bounds__(256)
void sage_gemm(const float* __restrict__ Aself,
               const float* __restrict__ Sneigh,   // segment sums [n_dst x K]
               const float* __restrict__ Cnt,      // edge counts  [n_dst]
               const float* __restrict__ Wself,    // [K x N] row-major
               const float* __restrict__ Wneigh,   // [K x N] row-major
               const float* __restrict__ Bias,     // [N]
               float* __restrict__ Out,            // [n_dst x N]
               int K, int N, int nblk, int tiles_m, int do_relu) {
    __shared__ __bf16 As[8][16][32];   // wave-private A-self tile  [m][k]
    __shared__ __bf16 An[8][16][32];   // wave-private A-neigh tile [m][k]
    __shared__ __bf16 Bs[64][32];      // block-shared W-self tile, transposed [n][k]
    __shared__ __bf16 Bn[64][32];      // block-shared W-neigh tile, transposed [n][k]
    __shared__ float  Inv[8][16];      // per-wave 1/max(cnt,1)

    const int tid   = threadIdx.x;
    const int lane  = tid & 31;
    const int wv    = tid >> 5;
    const int blk_m = blockIdx.x / nblk;
    const int blk_n = blockIdx.x - blk_m * nblk;
    const int row0b = blk_m * 128;            // block-uniform
    const int n0    = blk_n * 64;             // block-uniform
    const int K2    = K >> 1;
    const int N2    = N >> 1;

    const bool valid  = (blk_m * 8 + wv) < tiles_m;
    const int  rowoff = valid ? wv * 16 : 0;  // clamp dead waves onto row 0 tile

    // Block-uniform typed base pointers (scalar) + small int offsets
    const float2* Ab2  = (const float2*)(Aself  + (size_t)row0b * K);
    const float2* Sb2  = (const float2*)(Sneigh + (size_t)row0b * K);
    const float2* Wsb2 = (const float2*)(Wself + n0);
    const float2* Wnb2 = (const float2*)(Wneigh + n0);
    const float*  Ab   = (const float*)Ab2;
    const float*  Sb   = (const float*)Sb2;
    float*        Ob   = Out + (size_t)row0b * N + n0;

    if (lane < 16) {
        float c = Cnt[row0b + rowoff + lane];
        Inv[wv][lane] = 1.0f / fmaxf(c, 1.0f);
    }
    __builtin_amdgcn_wave_barrier();   // same-wave LDS producer/consumer

    v8f acc0 = {}, acc1 = {}, acc2 = {}, acc3 = {};

    const int nloc = lane & 15;          // A-row / B-column held by this lane
    const int kbA  = (lane >> 4) * 8;    // A frag: K in [kbA,kbA+8) U [kbA+16,kbA+24)
    const int kbB  = (lane >> 4) * 8;    // B frag: K in [kbB,kbB+8) U [kbB+16,kbB+24)
    const int lp   = lane & 15;          // A staging: column pair index (0..15)
    const int rh   = (lane >> 4) * 8;    // A staging: row half offset (0 or 8)

    for (int k0 = 0; k0 < K; k0 += 32) {
        __syncthreads();                 // previous k-step's consumers are done

        // ---- wave-private A staging: float2 loads, packed bf16x2 stores ----
#pragma unroll
        for (int r = 0; r < 8; ++r) {
            int row  = rh + r;
            int off2 = (rowoff + row) * K2 + (k0 >> 1) + lp;
            float2 a = Ab2[off2];
            float2 s = Sb2[off2];
            float iv = Inv[wv][row];
            v2bf pa = { (__bf16)a.x, (__bf16)a.y };
            v2bf pn = { (__bf16)(s.x * iv), (__bf16)(s.y * iv) };
            *(v2bf*)&As[wv][row][2 * lp] = pa;
            *(v2bf*)&An[wv][row][2 * lp] = pn;
        }
        // ---- block-cooperative B staging (32x64 each), float2 loads ----
#pragma unroll
        for (int i = 0; i < 4; ++i) {
            int p    = tid + 256 * i;    // pair index 0..1023
            int kk   = p >> 5;           // 0..31
            int np   = p & 31;           // column pair 0..31
            int off2 = (k0 + kk) * N2 + np;
            float2 w = Wsb2[off2];
            float2 u = Wnb2[off2];
            int nn = 2 * np;
            Bs[nn][kk]     = (__bf16)w.x;
            Bs[nn + 1][kk] = (__bf16)w.y;
            Bn[nn][kk]     = (__bf16)u.x;
            Bn[nn + 1][kk] = (__bf16)u.y;
        }

        // Prefetch next k-step's activation rows (global_prefetch_b8)
        if (k0 + 32 < K) {
            int poff = (rowoff + nloc) * K + k0 + 32 + (lane >> 4) * 16;
            __builtin_prefetch(&Ab[poff], 0, 0);
            __builtin_prefetch(&Sb[poff], 0, 0);
        }

        __syncthreads();

        // ---- fragments + 8 WMMAs ----
        v8bf aslo = *(const v8bf*)&As[wv][nloc][kbA];
        v8bf ashi = *(const v8bf*)&As[wv][nloc][kbA + 16];
        v16bf afs = __builtin_shufflevector(aslo, ashi, 0,1,2,3,4,5,6,7,8,9,10,11,12,13,14,15);
        v8bf anlo = *(const v8bf*)&An[wv][nloc][kbA];
        v8bf anhi = *(const v8bf*)&An[wv][nloc][kbA + 16];
        v16bf afn = __builtin_shufflevector(anlo, anhi, 0,1,2,3,4,5,6,7,8,9,10,11,12,13,14,15);

#pragma unroll
        for (int j = 0; j < 4; ++j) {
            v8bf bslo = *(const v8bf*)&Bs[16 * j + nloc][kbB];
            v8bf bshi = *(const v8bf*)&Bs[16 * j + nloc][kbB + 16];
            v16bf bfs = __builtin_shufflevector(bslo, bshi, 0,1,2,3,4,5,6,7,8,9,10,11,12,13,14,15);
            v8bf bnlo = *(const v8bf*)&Bn[16 * j + nloc][kbB];
            v8bf bnhi = *(const v8bf*)&Bn[16 * j + nloc][kbB + 16];
            v16bf bfn = __builtin_shufflevector(bnlo, bnhi, 0,1,2,3,4,5,6,7,8,9,10,11,12,13,14,15);
            v8f a = (j == 0) ? acc0 : (j == 1) ? acc1 : (j == 2) ? acc2 : acc3;
            a = __builtin_amdgcn_wmma_f32_16x16x32_bf16(false, afs, false, bfs,
                                                        (short)0, a, false, false);
            a = __builtin_amdgcn_wmma_f32_16x16x32_bf16(false, afn, false, bfn,
                                                        (short)0, a, false, false);
            if (j == 0) acc0 = a; else if (j == 1) acc1 = a;
            else if (j == 2) acc2 = a; else acc3 = a;
        }
    }

    // Epilogue. C/D layout: VGPR v -> M = v + 8*(lane>=16); N = lane&15.
    if (valid) {
        const int mofs = (lane >> 4) * 8;
        const float* Bb = Bias + n0;
#pragma unroll
        for (int j = 0; j < 4; ++j) {
            v8f a = (j == 0) ? acc0 : (j == 1) ? acc1 : (j == 2) ? acc2 : acc3;
            float bias = Bb[16 * j + nloc];
#pragma unroll
            for (int v = 0; v < 8; ++v) {
                float o = a[v] + bias;
                if (do_relu) o = fmaxf(o, 0.0f);
                Ob[(rowoff + mofs + v) * N + 16 * j + nloc] = o;
            }
        }
    }
}

static inline int idiv_up(int a, int b) { return (a + b - 1) / b; }

extern "C" void kernel_launch(void* const* d_in, const int* in_sizes, int n_in,
                              void* d_out, int out_size, void* d_ws, size_t ws_size,
                              hipStream_t stream) {
    (void)in_sizes; (void)n_in; (void)out_size; (void)ws_size;
    const int N1 = 100000, N2 = 20000, N3 = 4096;
    const int E0 = 1500000, E1 = 200000, E2 = 40960;
    const int HID = 256, OUTF = 128;

    const float* feat = (const float*)d_in[0];
    const int* e0s = (const int*)d_in[1];
    const int* e0d = (const int*)d_in[2];
    const int* e1s = (const int*)d_in[3];
    const int* e1d = (const int*)d_in[4];
    const int* e2s = (const int*)d_in[5];
    const int* e2d = (const int*)d_in[6];
    const float* ws0 = (const float*)d_in[7];
    const float* wn0 = (const float*)d_in[8];
    const float* b0  = (const float*)d_in[9];
    const float* ws1 = (const float*)d_in[10];
    const float* wn1 = (const float*)d_in[11];
    const float* b1  = (const float*)d_in[12];
    const float* ws2 = (const float*)d_in[13];
    const float* wn2 = (const float*)d_in[14];
    const float* b2  = (const float*)d_in[15];
    float* out = (float*)d_out;

    // Workspace layout (bytes): sum reused across layers (sized for layer 0)
    char* wsp  = (char*)d_ws;
    float* sum = (float*)(wsp);                    // 102,400,000 B
    float* cnt = (float*)(wsp + 102400000);        //     400,000 B
    float* h1  = (float*)(wsp + 102800000);        // 102,400,000 B
    float* h2  = (float*)(wsp + 205200000);        //  20,480,000 B

    // ---------------- Layer 0: feat -> h1 (N1 x 256), ReLU ----------------
    hipMemsetAsync(sum, 0, (size_t)N1 * HID * sizeof(float), stream);
    hipMemsetAsync(cnt, 0, (size_t)N1 * sizeof(float), stream);
    sage_scatter<<<idiv_up(E0, 8), 256, 0, stream>>>(feat, e0s, e0d, sum, cnt, E0);
    {
        int nblk = HID / 64, tiles_m = N1 / 16;
        int grid = idiv_up(tiles_m, 8) * nblk;
        sage_gemm<<<grid, 256, 0, stream>>>(
            feat, sum, cnt, ws0, wn0, b0, h1, HID, HID, nblk, tiles_m, 1);
    }

    // ---------------- Layer 1: h1 -> h2 (N2 x 256), ReLU ----------------
    hipMemsetAsync(sum, 0, (size_t)N2 * HID * sizeof(float), stream);
    hipMemsetAsync(cnt, 0, (size_t)N2 * sizeof(float), stream);
    sage_scatter<<<idiv_up(E1, 8), 256, 0, stream>>>(h1, e1s, e1d, sum, cnt, E1);
    {
        int nblk = HID / 64, tiles_m = N2 / 16;
        int grid = idiv_up(tiles_m, 8) * nblk;
        sage_gemm<<<grid, 256, 0, stream>>>(
            h1, sum, cnt, ws1, wn1, b1, h2, HID, HID, nblk, tiles_m, 1);
    }

    // ---------------- Layer 2: h2 -> out (N3 x 128), no act ----------------
    hipMemsetAsync(sum, 0, (size_t)N3 * HID * sizeof(float), stream);
    hipMemsetAsync(cnt, 0, (size_t)N3 * sizeof(float), stream);
    sage_scatter<<<idiv_up(E2, 8), 256, 0, stream>>>(h2, e2s, e2d, sum, cnt, E2);
    {
        int nblk = OUTF / 64, tiles_m = N3 / 16;
        int grid = idiv_up(tiles_m, 8) * nblk;
        sage_gemm<<<grid, 256, 0, stream>>>(
            h2, sum, cnt, ws2, wn2, b2, out, HID, OUTF, nblk, tiles_m, 0);
    }
}
